// VectorQuantizer_60069412602503
// MI455X (gfx1250) — matrix operand order repaired and checked by reading, not compiled, not run on hardware
//
#include <hip/hip_runtime.h>
#include <hip/hip_bf16.h>

#define DIM   256
#define KCB   4096
#define NLAT  65536
#define TILEK 64
#define LDSS  264   // padded row stride in ushorts: 528B -> 132 dwords, 132%64=4 -> conflict-free B reads
#define BUFSZ (TILEK * LDSS)

typedef __attribute__((ext_vector_type(16))) __bf16          v16bf;
typedef __attribute__((ext_vector_type(8)))  float           v8f;

static __device__ __forceinline__ unsigned short f2bf(float f) {
    unsigned int u = __float_as_uint(f);
    // round-to-nearest-even truncation to bf16
    unsigned int r = u + 0x7FFFu + ((u >> 16) & 1u);
    return (unsigned short)(r >> 16);
}

// ---------------------------------------------------------------------------
// Prep: E (f32) -> Ebf (bf16), per-code squared norms, zero loss accumulator.
// grid = KCB blocks, 256 threads (one block per codebook row).
// ---------------------------------------------------------------------------
__global__ __launch_bounds__(256) void vq_prep(const float* __restrict__ E,
                                               unsigned short* __restrict__ Ebf,
                                               float* __restrict__ En,
                                               float* __restrict__ acc) {
    __shared__ float red[256];
    const int k = blockIdx.x;
    const int d = threadIdx.x;
    float e = E[(size_t)k * DIM + d];
    Ebf[(size_t)k * DIM + d] = f2bf(e);
    red[d] = e * e;
    __syncthreads();
    for (int s = 128; s > 0; s >>= 1) {
        if (d < s) red[d] += red[d + s];
        __syncthreads();
    }
    if (d == 0) {
        En[k] = red[0];
        if (k == 0) *acc = 0.0f;
    }
}

// ---------------------------------------------------------------------------
// Argmin over K via bf16 WMMA. 8 waves/block; each wave owns 16 latent rows.
// Double-buffered LDS staging of 64-code tiles: global loads for tile k+1 are
// issued before the 32-WMMA compute burst on tile k; one barrier per tile.
// Score = ||e||^2 - 2*(x.e)  (||x||^2 is constant per row -> dropped).
// ---------------------------------------------------------------------------
__global__ __launch_bounds__(256) void vq_argmin(const float* __restrict__ X,
                                                 const unsigned short* __restrict__ Ebf,
                                                 const float* __restrict__ En,
                                                 int* __restrict__ ind) {
    __shared__ unsigned short sB[2 * BUFSZ];      // 2 x 33792 B
    __shared__ float sEn[2][TILEK];

    const int t       = threadIdx.x;
    const int lane    = t & 31;
    const int wave    = t >> 5;
    const int l15     = lane & 15;
    const int halfHi  = (lane >= 16) ? 1 : 0;
    const int rowBase = blockIdx.x * 128 + wave * 16;

    const int sr = t >> 2;          // staging row 0..63
    const int sq = t & 3;           // staging quarter of a row (64 ushorts)

    // ---- Load A tile (16 x 256) once, f32 -> bf16, ISA 16-bit A layout:
    // lane(l<16): row=l, elems j<8 -> d=32c+j, j>=8 -> d=32c+16+(j-8); lanes>=16: +8.
    v16bf a[8];
    {
        const float* xr = X + (size_t)(rowBase + l15) * DIM;
        const int hi = halfHi * 8;
        for (int c8 = 0; c8 < 8; ++c8) {
            const int db = c8 * 32 + hi;
            float4 f0 = *(const float4*)(xr + db);
            float4 f1 = *(const float4*)(xr + db + 4);
            float4 f2 = *(const float4*)(xr + db + 16);
            float4 f3 = *(const float4*)(xr + db + 20);
            union { v16bf v; unsigned short s[16]; } ua;
            ua.s[0] = f2bf(f0.x);  ua.s[1] = f2bf(f0.y);
            ua.s[2] = f2bf(f0.z);  ua.s[3] = f2bf(f0.w);
            ua.s[4] = f2bf(f1.x);  ua.s[5] = f2bf(f1.y);
            ua.s[6] = f2bf(f1.z);  ua.s[7] = f2bf(f1.w);
            ua.s[8]  = f2bf(f2.x); ua.s[9]  = f2bf(f2.y);
            ua.s[10] = f2bf(f2.z); ua.s[11] = f2bf(f2.w);
            ua.s[12] = f2bf(f3.x); ua.s[13] = f2bf(f3.y);
            ua.s[14] = f2bf(f3.z); ua.s[15] = f2bf(f3.w);
            a[c8] = ua.v;
        }
    }

    float minv[8];
    int   mini[8];
#pragma unroll
    for (int r = 0; r < 8; ++r) { minv[r] = 3.4e38f; mini[r] = 0; }

    // ---- Prologue: stage tile 0 into buffer 0
    {
        const uint4* src = (const uint4*)(Ebf + (size_t)sr * DIM + sq * 64);
        uint4* dst = (uint4*)(sB + sr * LDSS + sq * 64);
#pragma unroll
        for (int i = 0; i < 8; ++i) dst[i] = src[i];
        if (t < TILEK) sEn[0][t] = En[t];
    }
    __syncthreads();

    for (int kt = 0; kt < KCB; kt += TILEK) {
        const int cur = (kt / TILEK) & 1;
        const unsigned short* sBc = sB + cur * BUFSZ;
        const bool more = (kt + TILEK) < KCB;

        // ---- Issue global loads for the NEXT tile (latency hides under WMMAs)
        uint4 stg[8];
        float en_n = 0.0f;
        if (more) {
            const uint4* src = (const uint4*)(Ebf + (size_t)(kt + TILEK + sr) * DIM + sq * 64);
#pragma unroll
            for (int i = 0; i < 8; ++i) stg[i] = src[i];
            if (t < TILEK) en_n = En[kt + TILEK + t];
        }

        // ---- Compute: 4 column subtiles x 8 K-chunks of WMMA
#pragma unroll
        for (int kt2 = 0; kt2 < 4; ++kt2) {
            v8f c = {};
#pragma unroll
            for (int c8 = 0; c8 < 8; ++c8) {
                // B layout: col = lane&15 -> code row, elem j<8 -> d = 32*c8 + (lane<16?0:8) + j, j>=8 -> +16+(j-8)
                const unsigned short* bp =
                    sBc + (kt2 * 16 + l15) * LDSS + c8 * 32 + halfHi * 8;
                union { v16bf v; uint4 q[2]; } ub;
                ub.q[0] = ((const uint4*)bp)[0];
                ub.q[1] = ((const uint4*)bp)[2];
                c = __builtin_amdgcn_wmma_f32_16x16x32_bf16(
                        false, a[c8], false, ub.v, (short)0, c, false, false);
            }
            const int   kglob = kt + kt2 * 16 + l15;
            const float en    = sEn[cur][kt2 * 16 + l15];
#pragma unroll
            for (int r = 0; r < 8; ++r) {
                float s = en - 2.0f * c[r];
                if (s < minv[r]) { minv[r] = s; mini[r] = kglob; }
            }
        }

        // ---- Store staged data into the other buffer, then single barrier
        if (more) {
            uint4* dst = (uint4*)(sB + (cur ^ 1) * BUFSZ + sr * LDSS + sq * 64);
#pragma unroll
            for (int i = 0; i < 8; ++i) dst[i] = stg[i];
            if (t < TILEK) sEn[cur ^ 1][t] = en_n;
        }
        __syncthreads();
    }

    // ---- Reduce argmin across the 16 lanes holding each row (C layout halves)
#pragma unroll
    for (int r = 0; r < 8; ++r) {
        float v = minv[r];
        int   i = mini[r];
        for (int m = 8; m >= 1; m >>= 1) {
            float v2 = __shfl_xor(v, m, 32);
            int   i2 = __shfl_xor(i, m, 32);
            if (v2 < v || (v2 == v && i2 < i)) { v = v2; i = i2; }
        }
        if (l15 == 0) ind[rowBase + halfHi * 8 + r] = i;
    }
}

// ---------------------------------------------------------------------------
// Gather quantized rows (fp32 codebook), write out[0:N*D], accumulate sum((q-x)^2).
// grid = N*D / 2048 = 8192 blocks.
// ---------------------------------------------------------------------------
__global__ __launch_bounds__(256) void vq_finalize(const float* __restrict__ X,
                                                   const float* __restrict__ E,
                                                   const int* __restrict__ ind,
                                                   float* __restrict__ outQ,
                                                   float* __restrict__ acc) {
    __shared__ float red[256];
    const int t = threadIdx.x;
    const size_t base = (size_t)blockIdx.x * 2048;
    float s = 0.0f;
#pragma unroll
    for (int j = 0; j < 8; ++j) {
        size_t i = base + (size_t)j * 256 + t;
        int n = (int)(i >> 8);
        int d = (int)(i & 255);
        float q = E[(size_t)ind[n] * DIM + d];
        float x = X[i];
        outQ[i] = q;                 // quantized_st == quantized numerically
        float df = q - x;
        s += df * df;
    }
    red[t] = s;
    __syncthreads();
    for (int st = 128; st > 0; st >>= 1) {
        if (t < st) red[t] += red[t + st];
        __syncthreads();
    }
    if (t == 0) atomicAdd(acc, red[0]);
}

__global__ void vq_loss_k(const float* __restrict__ acc, float* __restrict__ out) {
    // commitment*BETA + embedding == (1 + 0.25) * mean((q-x)^2)
    out[0] = 1.25f * acc[0] / (float)(NLAT * DIM);
}

// ---------------------------------------------------------------------------
// One-hot fill: 1.07 GB of coalesced b32 stores (region is 4B-aligned only).
// grid = NLAT blocks.
// ---------------------------------------------------------------------------
__global__ __launch_bounds__(256) void vq_onehot(const int* __restrict__ ind,
                                                 float* __restrict__ outOH) {
    __shared__ int si;
    const int n = blockIdx.x;
    if (threadIdx.x == 0) si = ind[n];
    __syncthreads();
    float* row = outOH + (size_t)n * KCB;
    const int kind = si;
#pragma unroll
    for (int i = 0; i < 16; ++i) {
        int k = i * 256 + threadIdx.x;
        row[k] = (k == kind) ? 1.0f : 0.0f;
    }
}

extern "C" void kernel_launch(void* const* d_in, const int* in_sizes, int n_in,
                              void* d_out, int out_size, void* d_ws, size_t ws_size,
                              hipStream_t stream) {
    const float* X = (const float*)d_in[0];   // latents  [N, D]
    const float* E = (const float*)d_in[1];   // codebook [K, D]
    float* out = (float*)d_out;

    // workspace layout
    char* ws = (char*)d_ws;
    float*          acc  = (float*)ws;                       // 4 B
    float*          En   = (float*)(ws + 64);                // 16 KB
    int*            ind  = (int*)(ws + 16448);               // 256 KB
    unsigned short* Ebf  = (unsigned short*)(ws + 278592);   // 2 MB

    float* outQ  = out;                                   // [N*D]
    float* outL  = out + (size_t)NLAT * DIM;              // [1]
    float* outOH = out + (size_t)NLAT * DIM + 1;          // [N*K]

    vq_prep<<<KCB, 256, 0, stream>>>(E, Ebf, En, acc);
    vq_argmin<<<NLAT / 128, 256, 0, stream>>>(X, Ebf, En, ind);
    vq_finalize<<<(NLAT * (size_t)DIM) / 2048, 256, 0, stream>>>(X, E, ind, outQ, acc);
    vq_loss_k<<<1, 1, 0, stream>>>(acc, outL);
    vq_onehot<<<NLAT, 256, 0, stream>>>(ind, outOH);
}
